// SpatialAttention_20435454395195
// MI455X (gfx1250) — matrix operand order repaired and checked by reading, not compiled, not run on hardware
//
#include <hip/hip_runtime.h>
#include <hip/hip_bf16.h>
#include <cstdint>
#include <cstddef>

// ---------------------------------------------------------------------------
// Problem constants (fixed by the reference)
// ---------------------------------------------------------------------------
constexpr int Cc = 256;   // channels
constexpr int C8 = 32;    // C/8 (qk dim)
constexpr int NN = 4096;  // H*W
constexpr int Bt = 4;     // batch

typedef __attribute__((ext_vector_type(16))) __bf16 v16bf;
typedef __attribute__((ext_vector_type(8)))  float  v8f;

union BF16x16 {           // one WMMA bf16 operand (8 VGPRs / lane)
    v16bf  v;
    uint4  u[2];
    __bf16 h[16];
};

#define WMMA_BF16(a, b, c) \
    __builtin_amdgcn_wmma_f32_16x16x32_bf16(false, (a), false, (b), (short)0, (c), false, false)

// ---- 16-lane-group cross-lane reductions via ds_swizzle (xor masks < 16
//      stay inside each half-wave group, matching the C/D row layout) -------
template <int MASK>
__device__ __forceinline__ float xor_swz(float x) {
    int i = __builtin_bit_cast(int, x);
    int j = __builtin_amdgcn_ds_swizzle(i, (MASK << 10) | 0x1f); // group-of-32, xor=MASK, and=0x1f
    return __builtin_bit_cast(float, j);
}
__device__ __forceinline__ float redmax16(float v) {
    v = fmaxf(v, xor_swz<1>(v)); v = fmaxf(v, xor_swz<2>(v));
    v = fmaxf(v, xor_swz<4>(v)); v = fmaxf(v, xor_swz<8>(v));
    return v;
}
__device__ __forceinline__ float redsum16(float v) {
    v += xor_swz<1>(v); v += xor_swz<2>(v);
    v += xor_swz<4>(v); v += xor_swz<8>(v);
    return v;
}

// ---------------------------------------------------------------------------
// Kernel 1: fused 1x1-conv projections via WMMA.
//   y[320, N] = [Wb;Wc;Wd] (320x256) @ x (256xN) + bias, per batch.
//   rows 0..31  -> fbT  [B,N,32]  bf16 (query, row-major over C8)
//   rows 32..63 -> fcT  [B,N,32]  bf16 (key,   row-major over C8)
//   rows 64..319-> fd   [B,C,N]   bf16 (value, channel-major)
// One wave handles one 16-pixel tile across all 20 row tiles.
// ---------------------------------------------------------------------------
__global__ __launch_bounds__(128)
void proj_kernel(const float* __restrict__ x,
                 const float* __restrict__ Wb, const float* __restrict__ bb,
                 const float* __restrict__ Wc, const float* __restrict__ bc,
                 const float* __restrict__ Wd, const float* __restrict__ bd,
                 __bf16* __restrict__ fbT, __bf16* __restrict__ fcT,
                 __bf16* __restrict__ fd)
{
    const int lane = threadIdx.x & 31;
    const int wave = threadIdx.x >> 5;
    const int tid  = blockIdx.x * 4 + wave;      // 1024 pixel tiles total
    const int b    = tid >> 8;                   // 256 tiles per batch
    const int n0   = (tid & 255) << 4;
    const int col  = lane & 15;
    const int hl   = lane >> 4;
    const int n    = n0 + col;

    const float* xb = x + (size_t)b * Cc * NN;

    // B-operand (x) for all 8 K-blocks of 32: lane = pixel column,
    // slot e -> channel k = kb*32 + (e&7) + 16*(e>>3) + 8*hl  (dense bf16 B layout).
    BF16x16 Xb[8];
#pragma unroll
    for (int kb = 0; kb < 8; ++kb) {
#pragma unroll
        for (int e = 0; e < 16; ++e) {
            int k = kb * 32 + (e & 7) + 16 * (e >> 3) + 8 * hl;
            Xb[kb].h[e] = (__bf16)xb[(size_t)k * NN + n];
        }
    }

#pragma unroll 1
    for (int rt = 0; rt < 20; ++rt) {
        const int row0 = rt * 16;
        const float* Wsel; const float* bsel; int rbase;
        if (row0 < 32)       { Wsel = Wb; bsel = bb; rbase = row0; }
        else if (row0 < 64)  { Wsel = Wc; bsel = bc; rbase = row0 - 32; }
        else                 { Wsel = Wd; bsel = bd; rbase = row0 - 64; }

        // A-operand (weights): lane row = rbase+col,
        // slot e -> k = kb*32 + (e&7) + 16*(e>>3) + 8*hl  (bf16 A layout).
        const float* wrow = Wsel + (size_t)(rbase + col) * Cc;
        v8f acc = {};
#pragma unroll
        for (int kb = 0; kb < 8; ++kb) {
            const float* wp = wrow + kb * 32 + hl * 8;
            float4 wa = *(const float4*)(wp);
            float4 wb4 = *(const float4*)(wp + 4);
            float4 wc4 = *(const float4*)(wp + 16);
            float4 wd4 = *(const float4*)(wp + 20);
            BF16x16 A;
            A.h[0] = (__bf16)wa.x;  A.h[1] = (__bf16)wa.y;
            A.h[2] = (__bf16)wa.z;  A.h[3] = (__bf16)wa.w;
            A.h[4] = (__bf16)wb4.x; A.h[5] = (__bf16)wb4.y;
            A.h[6] = (__bf16)wb4.z; A.h[7] = (__bf16)wb4.w;
            A.h[8] = (__bf16)wc4.x; A.h[9] = (__bf16)wc4.y;
            A.h[10] = (__bf16)wc4.z; A.h[11] = (__bf16)wc4.w;
            A.h[12] = (__bf16)wd4.x; A.h[13] = (__bf16)wd4.y;
            A.h[14] = (__bf16)wd4.z; A.h[15] = (__bf16)wd4.w;
            acc = WMMA_BF16(A.v, Xb[kb].v, acc);
        }

        // C/D layout: vgpr r, lane -> (row = rbase + r + 8*hl, pixel n)
#pragma unroll
        for (int r = 0; r < 8; ++r) {
            int co = rbase + r + 8 * hl;
            float v = acc[r] + bsel[co];
            if (row0 < 32)
                fbT[((size_t)b * NN + n) * C8 + co] = (__bf16)v;
            else if (row0 < 64)
                fcT[((size_t)b * NN + n) * C8 + co] = (__bf16)v;
            else
                fd[((size_t)b * Cc + co) * NN + n] = (__bf16)v;
        }
    }
}

// ---------------------------------------------------------------------------
// Kernel 2: fused flash attention + alpha-residual.
// One wave owns 16 queries, accumulates all 256 output channels (16 v8f
// tiles), streams 32 keys/iter: 2 score WMMAs + online softmax + 16 PV WMMAs.
// P converts C-layout -> A-layout through a per-wave LDS slice.
// ---------------------------------------------------------------------------
__global__ __launch_bounds__(128)
void attn_kernel(const float* __restrict__ x, const float* __restrict__ alpha,
                 const __bf16* __restrict__ fbT, const __bf16* __restrict__ fcT,
                 const __bf16* __restrict__ fd, float* __restrict__ out)
{
    __shared__ alignas(16) __bf16 sP[4][16 * 32];

    const int lane = threadIdx.x & 31;
    const int wave = threadIdx.x >> 5;
    const int tid  = blockIdx.x * 4 + wave;
    const int b    = tid >> 8;
    const int n0   = (tid & 255) << 4;
    const int col  = lane & 15;
    const int hl   = lane >> 4;

    // Q A-operand (loop invariant): lane row = n0+col,
    // slots 0..15 = K 16*hl..16*hl+15 (contiguous k order, same as K)
    BF16x16 Qa;
    {
        const char* qrow = (const char*)(fbT + ((size_t)b * NN + n0 + col) * C8);
        Qa.u[0] = *(const uint4*)(qrow + 32 * hl);
        Qa.u[1] = *(const uint4*)(qrow + 16 + 32 * hl);
    }

    v8f O[16];
#pragma unroll
    for (int t = 0; t < 16; ++t) O[t] = v8f{};
    float mrow[8], lrow[8];
#pragma unroll
    for (int r = 0; r < 8; ++r) { mrow[r] = -__builtin_inff(); lrow[r] = 0.f; }

    __bf16* myP = sP[wave];
    const __bf16* fdB = fd + (size_t)b * Cc * NN;

#pragma unroll 1
    for (int j0 = 0; j0 < NN; j0 += 32) {
        // K B-operands for keys j0..j0+15 and j0+16..j0+31
        BF16x16 K0, K1;
        {
            const char* k0 = (const char*)(fcT + ((size_t)b * NN + j0 + col) * C8) + 32 * hl;
            K0.u[0] = *(const uint4*)k0;  K0.u[1] = *(const uint4*)(k0 + 16);
            const char* k1 = (const char*)(fcT + ((size_t)b * NN + j0 + 16 + col) * C8) + 32 * hl;
            K1.u[0] = *(const uint4*)k1;  K1.u[1] = *(const uint4*)(k1 + 16);
        }
        v8f s0 = {}, s1 = {};
        s0 = WMMA_BF16(Qa.v, K0.v, s0);   // scores, keys j0..+15
        s1 = WMMA_BF16(Qa.v, K1.v, s1);   // scores, keys j0+16..+31

        // Online softmax; C-layout: vgpr r, lane -> row r+8*hl, key col(+16)
        float scale[8];
#pragma unroll
        for (int r = 0; r < 8; ++r) {
            float mb = redmax16(fmaxf(s0[r], s1[r]));
            float mn = fmaxf(mrow[r], mb);
            float sc = __expf(mrow[r] - mn);
            float p0 = __expf(s0[r] - mn);
            float p1 = __expf(s1[r] - mn);
            float sb = redsum16(p0 + p1);
            lrow[r] = lrow[r] * sc + sb;
            mrow[r] = mn;
            scale[r] = sc;
            int row = r + 8 * hl;
            myP[row * 32 + col]      = (__bf16)p0;
            myP[row * 32 + 16 + col] = (__bf16)p1;
        }
#pragma unroll
        for (int t = 0; t < 16; ++t)
#pragma unroll
            for (int r = 0; r < 8; ++r)
                O[t][r] *= scale[r];

        // Re-read P in A layout (per-wave LDS slice, in-wave DS ordering)
        BF16x16 Pa;
        {
            const char* pr = (const char*)(myP + col * 32) + 32 * hl;
            Pa.u[0] = *(const uint4*)pr;
            Pa.u[1] = *(const uint4*)(pr + 16);
        }

        // PV: V B-tile = fd[c, j0+16*hl .. +15], contiguous 32B per lane
#pragma unroll
        for (int t = 0; t < 16; ++t) {
            BF16x16 Vb;
            const char* vr = (const char*)(fdB + (size_t)(t * 16 + col) * NN + j0 + 16 * hl);
            Vb.u[0] = *(const uint4*)vr;
            Vb.u[1] = *(const uint4*)(vr + 16);
            O[t] = WMMA_BF16(Pa.v, Vb.v, O[t]);
        }
    }

    // Epilogue: out = alpha * O/lsum + x. Rows r+8*hl are contiguous in q.
    const float a = alpha[0];
    float inv[8];
#pragma unroll
    for (int r = 0; r < 8; ++r) inv[r] = a / lrow[r];
    const float* xb = x + (size_t)b * Cc * NN;
    float* ob = out + (size_t)b * Cc * NN;
    const int qb = n0 + 8 * hl;
#pragma unroll
    for (int t = 0; t < 16; ++t) {
        int c = t * 16 + col;
        size_t base = (size_t)c * NN + qb;
        float4 x0 = *(const float4*)(xb + base);
        float4 x1 = *(const float4*)(xb + base + 4);
        float4 o0, o1;
        o0.x = fmaf(O[t][0], inv[0], x0.x);
        o0.y = fmaf(O[t][1], inv[1], x0.y);
        o0.z = fmaf(O[t][2], inv[2], x0.z);
        o0.w = fmaf(O[t][3], inv[3], x0.w);
        o1.x = fmaf(O[t][4], inv[4], x1.x);
        o1.y = fmaf(O[t][5], inv[5], x1.y);
        o1.z = fmaf(O[t][6], inv[6], x1.z);
        o1.w = fmaf(O[t][7], inv[7], x1.w);
        *(float4*)(ob + base)     = o0;
        *(float4*)(ob + base + 4) = o1;
    }
}

// ---------------------------------------------------------------------------
extern "C" void kernel_launch(void* const* d_in, const int* in_sizes, int n_in,
                              void* d_out, int out_size, void* d_ws, size_t ws_size,
                              hipStream_t stream)
{
    (void)in_sizes; (void)n_in; (void)out_size; (void)ws_size;
    const float* x     = (const float*)d_in[0];
    const float* Wb    = (const float*)d_in[1];
    const float* bb    = (const float*)d_in[2];
    const float* Wc    = (const float*)d_in[3];
    const float* bc    = (const float*)d_in[4];
    const float* Wd    = (const float*)d_in[5];
    const float* bd    = (const float*)d_in[6];
    const float* alpha = (const float*)d_in[7];

    char* ws = (char*)d_ws;
    const size_t fb_bytes = (size_t)Bt * NN * C8 * sizeof(__bf16);  // 1 MB
    __bf16* fbT = (__bf16*)ws;
    __bf16* fcT = (__bf16*)(ws + fb_bytes);
    __bf16* fd  = (__bf16*)(ws + 2 * fb_bytes);                     // 8 MB

    const int tiles  = Bt * NN / 16;   // 1024 waves of work
    const dim3 blk(128);               // 4 waves / block
    const dim3 grd(tiles / 4);         // 256 blocks

    proj_kernel<<<grd, blk, 0, stream>>>(x, Wb, bb, Wc, bc, Wd, bd, fbT, fcT, fd);
    attn_kernel<<<grd, blk, 0, stream>>>(x, alpha, fbT, fcT, fd, (float*)d_out);
}
